// VoxelWiseEncodingModel_25228637896911
// MI455X (gfx1250) — matrix operand order repaired and checked by reading, not compiled, not run on hardware
//
#include <hip/hip_runtime.h>

#define T_DIM 512
#define D_DIM 512
#define V_DIM 2048
#define H_DIM 128
#define TT    128     // T rows per tile
#define KC    128     // K chunk staged in LDS for layer 1
#define LDA   136     // padded LDS row stride in f16 elements (272B)

typedef _Float16 v2h  __attribute__((ext_vector_type(2)));
typedef _Float16 v4h  __attribute__((ext_vector_type(4)));
typedef _Float16 v8h  __attribute__((ext_vector_type(8)));
typedef _Float16 v16h __attribute__((ext_vector_type(16)));
typedef float    v4f  __attribute__((ext_vector_type(4)));
typedef float    v8f  __attribute__((ext_vector_type(8)));

__device__ __forceinline__ v16h cat8(v8h lo, v8h hi) {
  return __builtin_shufflevector(lo, hi, 0,1,2,3,4,5,6,7,8,9,10,11,12,13,14,15);
}

__device__ __forceinline__ v8h cat4(v4h lo, v4h hi) {
  return __builtin_shufflevector(lo, hi, 0,1,2,3,4,5,6,7);
}

// Stage a 128x128 fp32 block (row-major [k][n], row stride = H_DIM) into LDS
// transposed as f16: dst[n][k], packing (k, k+1) pairs into single b32 stores.
__device__ __forceinline__ void stage_transpose_f16(
    _Float16* __restrict__ dst, const float* __restrict__ src, int tid) {
  #pragma unroll
  for (int i = 0; i < 8; ++i) {
    int c = tid + 256 * i;          // 2048 blocks of 2(k) x 4(n)
    int kp = c >> 5;                // 0..63
    int n  = (c & 31) * 4;          // 0..124
    int k  = kp * 2;
    v4f f0 = *(const v4f*)(src + (size_t)(k + 0) * H_DIM + n);
    v4f f1 = *(const v4f*)(src + (size_t)(k + 1) * H_DIM + n);
    #pragma unroll
    for (int j = 0; j < 4; ++j) {
      v2h p;
      p.x = (_Float16)f0[j];
      p.y = (_Float16)f1[j];
      *(v2h*)(dst + (n + j) * LDA + k) = p;   // one ds_store_b32 per pair
    }
  }
}

__global__ __launch_bounds__(256) void voxel_mlp_kernel(
    const float* __restrict__ X,  const float* __restrict__ W1,
    const float* __restrict__ B1, const float* __restrict__ W2,
    const float* __restrict__ B2, const float* __restrict__ W3,
    const float* __restrict__ B3, float* __restrict__ out) {
  extern __shared__ char smem[];
  _Float16* As  = (_Float16*)smem;          // [TT][LDA] A tile (X), row-major f16
  _Float16* Bs  = As  + TT * LDA;           // [128][LDA] B tile, transposed (N-major, K contiguous)
  _Float16* H1s = Bs  + TT * LDA;           // [TT][LDA] layer-1 activations
  _Float16* H2s = H1s + TT * LDA;           // [TT][LDA] layer-2 activations
  float* sb1 = (float*)(H2s + TT * LDA);
  float* sb2 = sb1 + H_DIM;
  float* sw3 = sb2 + H_DIM;

  const int v    = blockIdx.x;
  const int tid  = threadIdx.x;
  const int wave = tid >> 5;
  const int lane = tid & 31;
  const int lrow = lane & 15;   // row/col within 16x16 tile
  const int ahalf = lane >> 4;  // which K-half this lane holds
  const int m0   = wave * 16;   // 16-row strip per wave

  const float* w1v = W1 + (size_t)v * D_DIM * H_DIM;
  const float* w2v = W2 + (size_t)v * H_DIM * H_DIM;

  if (tid < H_DIM) {
    sb1[tid] = B1[(size_t)v * H_DIM + tid];
    sb2[tid] = B2[(size_t)v * H_DIM + tid];
    sw3[tid] = W3[(size_t)v * H_DIM + tid];
  }
  const float bias3 = B3[v];

  const v8f vzero = {0.f,0.f,0.f,0.f,0.f,0.f,0.f,0.f};

  #pragma unroll 1
  for (int tt = 0; tt < T_DIM / TT; ++tt) {
    const int t0 = tt * TT;

    // ----------------- Layer 1: H1 = relu(X @ W1[v] + b1) -----------------
    v8f acc[8];
    #pragma unroll
    for (int n = 0; n < 8; ++n) acc[n] = vzero;

    #pragma unroll 1
    for (int kc = 0; kc < D_DIM / KC; ++kc) {
      const int k0 = kc * KC;
      __syncthreads();
      // stage X[t0:t0+128, k0:k0+128] -> As (f16, row-major), 8 elems/store
      #pragma unroll
      for (int i = 0; i < 8; ++i) {
        int c = tid + 256 * i;        // 2048 chunks of 8 along k
        int t = c >> 4;
        int k = (c & 15) * 8;
        const float* sp = X + (size_t)(t0 + t) * D_DIM + (k0 + k);
        v4f f0 = *(const v4f*)(sp);
        v4f f1 = *(const v4f*)(sp + 4);
        *(v8h*)(As + t * LDA + k) =
            cat4(__builtin_convertvector(f0, v4h),
                 __builtin_convertvector(f1, v4h));   // one ds_store_b128
      }
      // stage W1[v, k0:k0+128, :] -> Bs transposed: Bs[h][k] (packed b32 stores)
      stage_transpose_f16(Bs, w1v + (size_t)k0 * H_DIM, tid);
      __syncthreads();
      #pragma unroll
      for (int ks = 0; ks < KC / 32; ++ks) {
        const int kk = ks * 32;
        const _Float16* ap = As + (m0 + lrow) * LDA + kk + ahalf * 16;
        v16h a = cat8(*(const v8h*)ap, *(const v8h*)(ap + 8));
        #pragma unroll
        for (int n = 0; n < 8; ++n) {
          const _Float16* bp = Bs + (n * 16 + lrow) * LDA + kk + ahalf * 16;
          v16h b = cat8(*(const v8h*)bp, *(const v8h*)(bp + 8));
          acc[n] = __builtin_amdgcn_wmma_f32_16x16x32_f16(
              false, a, false, b, (short)0, acc[n], false, false);
        }
      }
    }
    // epilogue: bias + relu -> H1s (f16). C layout: VGPR r = row r (+8 for hi lanes)
    #pragma unroll
    for (int n = 0; n < 8; ++n) {
      float bias = sb1[n * 16 + lrow];
      #pragma unroll
      for (int r = 0; r < 8; ++r) {
        float x = fmaxf(acc[n][r] + bias, 0.f);
        H1s[(m0 + ahalf * 8 + r) * LDA + n * 16 + lrow] = (_Float16)x;
      }
    }
    __syncthreads();

    // ----------------- Layer 2: H2 = relu(H1 @ W2[v] + b2) ----------------
    // stage W2[v, h, g] -> Bs transposed: Bs[g][h]
    stage_transpose_f16(Bs, w2v, tid);
    __syncthreads();

    v8f acc2[8];
    #pragma unroll
    for (int n = 0; n < 8; ++n) acc2[n] = vzero;
    #pragma unroll
    for (int ks = 0; ks < H_DIM / 32; ++ks) {
      const int kk = ks * 32;
      const _Float16* ap = H1s + (m0 + lrow) * LDA + kk + ahalf * 16;
      v16h a = cat8(*(const v8h*)ap, *(const v8h*)(ap + 8));
      #pragma unroll
      for (int n = 0; n < 8; ++n) {
        const _Float16* bp = Bs + (n * 16 + lrow) * LDA + kk + ahalf * 16;
        v16h b = cat8(*(const v8h*)bp, *(const v8h*)(bp + 8));
        acc2[n] = __builtin_amdgcn_wmma_f32_16x16x32_f16(
            false, a, false, b, (short)0, acc2[n], false, false);
      }
    }
    #pragma unroll
    for (int n = 0; n < 8; ++n) {
      float bias = sb2[n * 16 + lrow];
      #pragma unroll
      for (int r = 0; r < 8; ++r) {
        float x = fmaxf(acc2[n][r] + bias, 0.f);
        H2s[(m0 + ahalf * 8 + r) * LDA + n * 16 + lrow] = (_Float16)x;
      }
    }
    __syncthreads();

    // ----------------- Layer 3: out[t,v] = H2[t,:] . W3[v] + b3 -----------
    if (tid < TT) {
      float sum = bias3;
      #pragma unroll
      for (int hc = 0; hc < H_DIM; hc += 8) {
        v8h hv = *(const v8h*)(H2s + tid * LDA + hc);   // one ds_load_b128
        #pragma unroll
        for (int j = 0; j < 8; ++j) sum += (float)hv[j] * sw3[hc + j];
      }
      out[(size_t)(t0 + tid) * V_DIM + v] = sum;
    }
    __syncthreads();
  }
}

extern "C" void kernel_launch(void* const* d_in, const int* in_sizes, int n_in,
                              void* d_out, int out_size, void* d_ws, size_t ws_size,
                              hipStream_t stream) {
  const float* X  = (const float*)d_in[0];
  const float* W1 = (const float*)d_in[1];
  const float* B1 = (const float*)d_in[2];
  const float* W2 = (const float*)d_in[3];
  const float* B2 = (const float*)d_in[4];
  const float* W3 = (const float*)d_in[5];
  const float* B3 = (const float*)d_in[6];
  float* out = (float*)d_out;

  const size_t lds_bytes = (size_t)4 * TT * LDA * sizeof(_Float16) + 3 * H_DIM * sizeof(float);
  voxel_mlp_kernel<<<dim3(V_DIM), dim3(256), lds_bytes, stream>>>(
      X, W1, B1, W2, B2, W3, B3, out);
}
